// LSTMAimModel_86251533238437
// MI455X (gfx1250) — matrix operand order repaired and checked by reading, not compiled, not run on hardware
//
#include <hip/hip_runtime.h>
#include <cstdint>
#include <cstddef>

// Problem constants (match reference)
#define B_   512
#define S_   64
#define D_   128
#define H_   1024
#define G4_  4096
#define T_   8
#define OUT_ 16
#define BH_  ((size_t)B_ * H_)

// Tiling
#define TM 64        // batch rows per workgroup
#define TN 64        // h-columns per workgroup (per gate; x4 gates computed)
#define TK 32        // K chunk (one bf16 WMMA depth)
#define APITCH 40    // 32 + 8 pad; 80-byte rows keep 16B alignment

#define AS3 __attribute__((address_space(3)))
#define AS1 __attribute__((address_space(1)))

#if defined(__has_builtin)
#if __has_builtin(__builtin_amdgcn_global_load_async_to_lds_b128) && \
    __has_builtin(__builtin_amdgcn_s_wait_asynccnt)
#define USE_ASYNC_LDS 1
#endif
#endif

typedef __attribute__((ext_vector_type(16))) __bf16         v16bf;
typedef __attribute__((ext_vector_type(8)))  float          v8f;
typedef __attribute__((ext_vector_type(8)))  unsigned short us8;
typedef __attribute__((ext_vector_type(4)))  int            v4i;

union Frag { v16bf v; us8 h[2]; };

__device__ __forceinline__ unsigned short f2bf(float f) {
  union { float f; uint32_t u; } v; v.f = f;
  uint32_t r = v.u + 0x7FFFu + ((v.u >> 16) & 1u);   // round-to-nearest-even
  return (unsigned short)(r >> 16);
}
__device__ __forceinline__ float bf2f(unsigned short s) {
  union { uint32_t u; float f; } v; v.u = ((uint32_t)s) << 16;
  return v.f;
}
__device__ __forceinline__ float sigf(float x) { return 1.0f / (1.0f + __expf(-x)); }

// ---------------------------------------------------------------------------
// Per-chunk WMMA compute.  8 waves arranged 2(M) x 4(N) over the 64x64 tile:
// wave tile = 32(M) x 16(N) per gate -> each B fragment feeds 2 WMMAs.
// AS3 pointers so fragment reads lower to ds_load_b128.
// ---------------------------------------------------------------------------
__device__ __forceinline__ void compute_chunk(
    const AS3 unsigned short* As,
    const AS3 unsigned short* Bs,
    v8f acc[4][2])
{
  const int lane = threadIdx.x & 31;
  const int wave = threadIdx.x >> 5;
  const int mw   = (wave & 1) * 32;    // 2 M-waves
  const int nw   = (wave >> 1) * 16;   // 4 N-waves
  const int am   = lane & 15;
  const int aK   = (lane < 16) ? 0 : 8;   // A frag: K={aK..+7} U {aK+16..+23}
  const int bK   = (lane < 16) ? 0 : 8;   // B frag: K={bK..+7} U {bK+16..+23}

  Frag a0, a1;
  a0.h[0] = *(const AS3 us8*)(As + (mw + am) * APITCH + aK);
  a0.h[1] = *(const AS3 us8*)(As + (mw + am) * APITCH + aK + 16);
  a1.h[0] = *(const AS3 us8*)(As + (mw + 16 + am) * APITCH + aK);
  a1.h[1] = *(const AS3 us8*)(As + (mw + 16 + am) * APITCH + aK + 16);

  const int n = nw + am;
#pragma unroll
  for (int g = 0; g < 4; ++g) {
    Frag b;
    b.h[0] = *(const AS3 us8*)(Bs + (g * TN + n) * APITCH + bK);
    b.h[1] = *(const AS3 us8*)(Bs + (g * TN + n) * APITCH + bK + 16);
    acc[g][0] = __builtin_amdgcn_wmma_f32_16x16x32_bf16(
        false, a0.v, false, b.v, (short)0, acc[g][0], false, false);
    acc[g][1] = __builtin_amdgcn_wmma_f32_16x16x32_bf16(
        false, a1.v, false, b.v, (short)0, acc[g][1], false, false);
  }
}

#if defined(USE_ASYNC_LDS)
// Issue one K-chunk of async global->LDS copies (5 x b128 per thread:
// 1 for the A tile, 4 for the four gate slices of W).  Tracked by ASYNCcnt.
__device__ __forceinline__ void stage_async(
    const unsigned short* __restrict__ A, int lda,
    const unsigned short* __restrict__ Bw, int ldb,
    int m0, int nh0, int k0,
    AS3 unsigned short* As, AS3 unsigned short* Bs)
{
  const int tid = threadIdx.x;
  const int row = tid >> 2;
  const int kc  = (tid & 3) * 8;
  __builtin_amdgcn_global_load_async_to_lds_b128(
      (AS1 v4i*)(uintptr_t)(A + (size_t)(m0 + row) * lda + (k0 + kc)),
      (AS3 v4i*)(As + row * APITCH + kc), 0, 0);
#pragma unroll
  for (int i = 0; i < 4; ++i) {
    int c  = tid + i * 256;
    int g  = c >> 8;
    int rr = (c & 255) >> 2;
    int bkc = (c & 3) * 8;
    __builtin_amdgcn_global_load_async_to_lds_b128(
        (AS1 v4i*)(uintptr_t)(Bw + (size_t)(g * H_ + nh0 + rr) * ldb + (k0 + bkc)),
        (AS3 v4i*)(Bs + (g * TN + rr) * APITCH + bkc), 0, 0);
  }
}
#else
// Fallback: synchronous staging through VGPRs.
__device__ __forceinline__ void stage_sync(
    const unsigned short* __restrict__ A, int lda,
    const unsigned short* __restrict__ Bw, int ldb,
    int m0, int nh0, int k0,
    AS3 unsigned short* As, AS3 unsigned short* Bs)
{
  const int tid = threadIdx.x;
  const int row = tid >> 2;
  const int kc  = (tid & 3) * 8;
  {
    const unsigned short* gp = A + (size_t)(m0 + row) * lda + (k0 + kc);
    us8 d = *(const us8*)gp;
    __builtin_prefetch(gp + TK, 0, 0);
    *(AS3 us8*)(As + row * APITCH + kc) = d;
  }
#pragma unroll
  for (int i = 0; i < 4; ++i) {
    int c  = tid + i * 256;
    int g  = c >> 8;
    int rr = (c & 255) >> 2;
    int bkc = (c & 3) * 8;
    const unsigned short* gp = Bw + (size_t)(g * H_ + nh0 + rr) * ldb + (k0 + bkc);
    us8 d = *(const us8*)gp;
    __builtin_prefetch(gp + TK, 0, 0);
    *(AS3 us8*)(Bs + (g * TN + rr) * APITCH + bkc) = d;
  }
}
#endif

// ---------------------------------------------------------------------------
// One K-segment of the fused 4-gate GEMM (double-buffered, async pipelined).
// A: [B_, K] bf16 (stride lda).  Bw: [4*H_, K] bf16 (stride ldb).
// ---------------------------------------------------------------------------
__device__ __forceinline__ void gemm_segment(
    const unsigned short* __restrict__ A, int lda,
    const unsigned short* __restrict__ Bw, int ldb, int K,
    int m0, int nh0,
    AS3 unsigned short* As0, AS3 unsigned short* Bs0,
    AS3 unsigned short* As1, AS3 unsigned short* Bs1,
    v8f acc[4][2])
{
  AS3 unsigned short* Ab[2] = {As0, As1};
  AS3 unsigned short* Bb[2] = {Bs0, Bs1};
  const int nch = K / TK;
#if defined(USE_ASYNC_LDS)
  stage_async(A, lda, Bw, ldb, m0, nh0, 0, Ab[0], Bb[0]);
  // Steady state: stage chunk i+1 while computing chunk i.
#pragma unroll 2
  for (int i = 0; i < nch - 1; ++i) {
    stage_async(A, lda, Bw, ldb, m0, nh0, (i + 1) * TK, Ab[(i + 1) & 1], Bb[(i + 1) & 1]);
    // 5 newest (next chunk) may remain outstanding; chunk i is complete
    // (async loads complete in order).
    __builtin_amdgcn_s_wait_asynccnt(5);
    __syncthreads();                 // everyone's chunk-i data visible in LDS
    compute_chunk(Ab[i & 1], Bb[i & 1], acc);
    __syncthreads();                 // done reading before buffer is reused
  }
  // Drain: last chunk.
  __builtin_amdgcn_s_wait_asynccnt(0);
  __syncthreads();
  compute_chunk(Ab[(nch - 1) & 1], Bb[(nch - 1) & 1], acc);
  __syncthreads();
#else
  for (int i = 0; i < nch; ++i) {
    __syncthreads();
    stage_sync(A, lda, Bw, ldb, m0, nh0, i * TK, Ab[0], Bb[0]);
    __syncthreads();
    compute_chunk(Ab[0], Bb[0], acc);
  }
  (void)As1; (void)Bs1;
#endif
}

// ---------------------------------------------------------------------------
// Fused LSTM step: G = A1@B1^T + A2@B2^T (+bias), gate nonlinearities,
// c update (in place, f32), h output (bf16).
// ---------------------------------------------------------------------------
__global__ __launch_bounds__(256) void lstm_step_kernel(
    const unsigned short* __restrict__ A1, int lda1,
    const unsigned short* __restrict__ B1, int ldb1, int K1,
    const unsigned short* __restrict__ A2, int lda2,
    const unsigned short* __restrict__ B2, int ldb2, int K2,
    const float* __restrict__ bias,
    float* __restrict__ cbuf,
    unsigned short* __restrict__ hout)
{
  __shared__ unsigned short As0[TM * APITCH];
  __shared__ unsigned short Bs0[4 * TN * APITCH];
  __shared__ unsigned short As1[TM * APITCH];
  __shared__ unsigned short Bs1[4 * TN * APITCH];

  // One addrspacecast each, hoisted out of all loops.
  AS3 unsigned short* As0p = (AS3 unsigned short*)As0;
  AS3 unsigned short* Bs0p = (AS3 unsigned short*)Bs0;
  AS3 unsigned short* As1p = (AS3 unsigned short*)As1;
  AS3 unsigned short* Bs1p = (AS3 unsigned short*)Bs1;

  const int m0  = blockIdx.x * TM;
  const int nh0 = blockIdx.y * TN;

  v8f acc[4][2];
#pragma unroll
  for (int g = 0; g < 4; ++g)
#pragma unroll
    for (int mf = 0; mf < 2; ++mf)
#pragma unroll
      for (int r = 0; r < 8; ++r) acc[g][mf][r] = 0.0f;

  gemm_segment(A1, lda1, B1, ldb1, K1, m0, nh0, As0p, Bs0p, As1p, Bs1p, acc); // h@Whh^T
  gemm_segment(A2, lda2, B2, ldb2, K2, m0, nh0, As0p, Bs0p, As1p, Bs1p, acc); // x@Wih^T

  // ---- epilogue: gates -> c,h.  C/D layout: (VGPR r, lane l) ->
  //      M = r + 8*(l>=16), N = l&15.  n is constant per lane here. ----
  const int lane = threadIdx.x & 31;
  const int wave = threadIdx.x >> 5;
  const int mw   = (wave & 1) * 32;
  const int nw   = (wave >> 1) * 16;
  const int mhi  = (lane >> 4) << 3;
  const int n    = nh0 + nw + (lane & 15);

  const float bi = bias[n];
  const float bf = bias[H_ + n];
  const float bg = bias[2 * H_ + n];
  const float bo = bias[3 * H_ + n];

#pragma unroll
  for (int mf = 0; mf < 2; ++mf) {
#pragma unroll
    for (int r = 0; r < 8; ++r) {
      const int m = m0 + mw + mf * 16 + r + mhi;
      const size_t idx = (size_t)m * H_ + n;
      const float ig = sigf(acc[0][mf][r] + bi);
      const float fg = sigf(acc[1][mf][r] + bf);
      const float gg = tanhf(acc[2][mf][r] + bg);
      const float og = sigf(acc[3][mf][r] + bo);
      const float c  = fg * cbuf[idx] + ig * gg;
      cbuf[idx] = c;
      hout[idx] = f2bf(og * tanhf(c));
    }
  }
}

// ---------------------------------------------------------------------------
// Utility kernels
// ---------------------------------------------------------------------------
__global__ void cvt_f32_bf16(const float* __restrict__ s,
                             unsigned short* __restrict__ d, int n) {
  int i = blockIdx.x * blockDim.x + threadIdx.x;
  int st = gridDim.x * blockDim.x;
  for (; i < n; i += st) d[i] = f2bf(s[i]);
}
__global__ void zero_f(float* __restrict__ p, int n) {
  int i = blockIdx.x * blockDim.x + threadIdx.x;
  int st = gridDim.x * blockDim.x;
  for (; i < n; i += st) p[i] = 0.0f;
}
__global__ void zero_h(unsigned short* __restrict__ p, int n) {
  int i = blockIdx.x * blockDim.x + threadIdx.x;
  int st = gridDim.x * blockDim.x;
  for (; i < n; i += st) p[i] = 0;
}
// out = dec_out @ W_out^T + b_out, duplicated to [B, 2*OUT]
__global__ void head_kernel(const unsigned short* __restrict__ hdec,
                            const float* __restrict__ Wout,
                            const float* __restrict__ bout,
                            float* __restrict__ out) {
  int tid = blockIdx.x * blockDim.x + threadIdx.x;  // 8192 = 512*16
  int b = tid >> 4, o = tid & 15;
  float s = bout[o];
  const unsigned short* hp = hdec + (size_t)b * H_;
  const float* wp = Wout + (size_t)o * H_;
  for (int k = 0; k < H_; ++k) s += bf2f(hp[k]) * wp[k];
  out[(size_t)b * 2 * OUT_ + o] = s;
  out[(size_t)b * 2 * OUT_ + OUT_ + o] = s;
}

// ---------------------------------------------------------------------------
extern "C" void kernel_launch(void* const* d_in, const int* in_sizes, int n_in,
                              void* d_out, int out_size, void* d_ws, size_t ws_size,
                              hipStream_t stream) {
  const float* x     = (const float*)d_in[0];
  const float* Wih0  = (const float*)d_in[1];
  const float* Whh0  = (const float*)d_in[2];
  const float* b0    = (const float*)d_in[3];
  const float* Wih1  = (const float*)d_in[4];
  const float* Whh1  = (const float*)d_in[5];
  const float* b1    = (const float*)d_in[6];
  const float* dWih  = (const float*)d_in[7];   // [2,4H,H]
  const float* dWhh  = (const float*)d_in[8];
  const float* db    = (const float*)d_in[9];   // [2,4H]
  const float* Wout  = (const float*)d_in[10];
  const float* bout  = (const float*)d_in[11];
  float* out = (float*)d_out;
  (void)in_sizes; (void)n_in; (void)out_size; (void)ws_size;

  // Workspace carve-up (~147 MB)
  char* ws = (char*)d_ws;
  size_t off = 0;
  auto carve = [&](size_t bytes) -> char* {
    char* p = ws + off;
    off += (bytes + 255) & ~(size_t)255;
    return p;
  };
  unsigned short* xbf    = (unsigned short*)carve((size_t)B_ * S_ * D_ * 2);
  unsigned short* Wih0b  = (unsigned short*)carve((size_t)G4_ * D_ * 2);
  unsigned short* Whh0b  = (unsigned short*)carve((size_t)G4_ * H_ * 2);
  unsigned short* Wih1b  = (unsigned short*)carve((size_t)G4_ * H_ * 2);
  unsigned short* Whh1b  = (unsigned short*)carve((size_t)G4_ * H_ * 2);
  unsigned short* dWihb  = (unsigned short*)carve((size_t)2 * G4_ * H_ * 2);
  unsigned short* dWhhb  = (unsigned short*)carve((size_t)2 * G4_ * H_ * 2);
  unsigned short* ys0    = (unsigned short*)carve((size_t)S_ * BH_ * 2);
  unsigned short* h1a    = (unsigned short*)carve(BH_ * 2);
  unsigned short* h1b    = (unsigned short*)carve(BH_ * 2);
  unsigned short* hd0a   = (unsigned short*)carve(BH_ * 2);
  unsigned short* hd0b   = (unsigned short*)carve(BH_ * 2);
  unsigned short* hd1a   = (unsigned short*)carve(BH_ * 2);
  unsigned short* hd1b   = (unsigned short*)carve(BH_ * 2);
  unsigned short* hzero  = (unsigned short*)carve(BH_ * 2);
  float* c0 = (float*)carve(BH_ * 4);
  float* c1 = (float*)carve(BH_ * 4);

  auto cvt = [&](const float* s, unsigned short* d, size_t n) {
    int blocks = (int)((n + 255) / 256); if (blocks > 2048) blocks = 2048;
    cvt_f32_bf16<<<blocks, 256, 0, stream>>>(s, d, (int)n);
  };
  cvt(x,    xbf,   (size_t)B_ * S_ * D_);
  cvt(Wih0, Wih0b, (size_t)G4_ * D_);
  cvt(Whh0, Whh0b, (size_t)G4_ * H_);
  cvt(Wih1, Wih1b, (size_t)G4_ * H_);
  cvt(Whh1, Whh1b, (size_t)G4_ * H_);
  cvt(dWih, dWihb, (size_t)2 * G4_ * H_);
  cvt(dWhh, dWhhb, (size_t)2 * G4_ * H_);
  zero_f<<<1024, 256, 0, stream>>>(c0, (int)BH_);
  zero_f<<<1024, 256, 0, stream>>>(c1, (int)BH_);
  zero_h<<<1024, 256, 0, stream>>>(hzero, (int)BH_);

  const dim3 sgrid(B_ / TM, H_ / TN);   // 8 x 16 workgroups
  auto step = [&](const unsigned short* A1, int lda1,
                  const unsigned short* Bw1, int ldb1, int K1,
                  const unsigned short* A2, int lda2,
                  const unsigned short* Bw2, int ldb2, int K2,
                  const float* bias, float* cb, unsigned short* ho) {
    lstm_step_kernel<<<sgrid, 256, 0, stream>>>(A1, lda1, Bw1, ldb1, K1,
                                                A2, lda2, Bw2, ldb2, K2,
                                                bias, cb, ho);
  };

  // ---- encoder layer 0: h history goes straight into ys0 ----
  for (int t = 0; t < S_; ++t) {
    const unsigned short* hp = t ? (ys0 + (size_t)(t - 1) * BH_) : hzero;
    step(hp, H_, Whh0b, H_, H_,
         xbf + (size_t)t * D_, S_ * D_, Wih0b, D_, D_,   // x[b,t,:] rows stride S*D
         b0, c0, ys0 + (size_t)t * BH_);
  }
  // ---- encoder layer 1: ping-pong h ----
  unsigned short* h1buf[2] = {h1a, h1b};
  for (int t = 0; t < S_; ++t) {
    const unsigned short* hp = t ? h1buf[(t - 1) & 1] : hzero;
    step(hp, H_, Whh1b, H_, H_,
         ys0 + (size_t)t * BH_, H_, Wih1b, H_, H_,
         b1, c1, h1buf[t & 1]);
  }
  unsigned short* h1fin = h1buf[(S_ - 1) & 1];

  // ---- autoregressive decoder (T_ steps, 2 layers, c continues in place) ----
  unsigned short* hd0buf[2] = {hd0a, hd0b};
  unsigned short* hd1buf[2] = {hd1a, hd1b};
  const unsigned short* hd0r = ys0 + (size_t)(S_ - 1) * BH_;  // enc L0 final h
  const unsigned short* hd1r = h1fin;                          // enc L1 final h
  const unsigned short* xin  = h1fin;                          // ys1[-1]
  for (int t = 0; t < T_; ++t) {
    unsigned short* hd0w = hd0buf[t & 1];
    unsigned short* hd1w = hd1buf[t & 1];
    step(hd0r, H_, dWhhb, H_, H_,
         xin,  H_, dWihb, H_, H_,
         db, c0, hd0w);
    step(hd1r, H_, dWhhb + (size_t)G4_ * H_, H_, H_,
         hd0w, H_, dWihb + (size_t)G4_ * H_, H_, H_,
         db + G4_, c1, hd1w);
    hd0r = hd0w; hd1r = hd1w; xin = hd1w;
  }

  head_kernel<<<(B_ * OUT_) / 256, 256, 0, stream>>>(
      hd1buf[(T_ - 1) & 1], Wout, bout, out);
}
